// SpectralTokeniser_29970281791979
// MI455X (gfx1250) — compile-verified, matrix operand order checked by reading, not run on hardware
//
#include <hip/hip_runtime.h>
#include <hip/hip_bf16.h>
#include <cmath>
#include <complex>

// ---------------- problem constants ----------------
#define NB 3
#define B_  128
#define C_  22
#define T_  5001
#define D_  64
#define L_  162
#define WIN 62
#define STR 31
#define TPAD 5120                 // 31 left pad + 5001 + right pad, room for l*31+w<=5054
#define MT  (B_ * L_)             // 20736 GEMM rows
#define KCH 44                    // K chunks of 32 (per-channel padded to 64: 22*64=1408)
#define BN_EPS 1e-5f

// chunk-parallel IIR: payload 512 samples, 512-sample zero-state warm-up.
// slowest pole radius (theta band) r~0.981 -> leftover transient r^512 ~ 5e-5,
// below the f16 quantization (~4e-4) already accepted in the WMMA A operand.
#define PCHUNK 512
#define WARM   512
#define NCHK   ((T_ + PCHUNK - 1) / PCHUNK)   // 10

typedef __attribute__((ext_vector_type(16))) _Float16 v16h;
typedef __attribute__((ext_vector_type(8)))  _Float16 v8h;
typedef v8h __attribute__((aligned(2)))      v8h_u;    // 2-byte aligned loads (l*31 offsets)
typedef __attribute__((ext_vector_type(8)))  float    v8f;

struct Sos  { float b0[4]; float b2[4]; float a1[4]; float a2[4]; };
struct Sos3 { Sos s[NB]; };

// ---------------- host: Butterworth bandpass SOS (exact port of reference) ----------------
static void butter_sos(double lo, double hi, Sos* out) {
  const double fs = 250.0, pi = 3.14159265358979323846;
  double w1 = 4.0 * std::tan(pi * (lo / (fs * 0.5)) * 0.5);
  double w2 = 4.0 * std::tan(pi * (hi / (fs * 0.5)) * 0.5);
  double bw = w2 - w1, wo = std::sqrt(w1 * w2);
  std::complex<double> pbp[8];
  for (int k = 1; k <= 4; ++k) {
    std::complex<double> plp = std::exp(std::complex<double>(0.0, pi * (2.0 * k + 3.0) / 8.0));
    std::complex<double> ps  = plp * (bw * 0.5);
    std::complex<double> rt  = std::sqrt(ps * ps - wo * wo);
    pbp[k - 1]     = ps + rt;
    pbp[k - 1 + 4] = ps - rt;
  }
  double kbp = bw * bw * bw * bw;
  std::complex<double> prod(1.0, 0.0);
  for (int i = 0; i < 8; ++i) prod *= (4.0 - pbp[i]);
  double kd = kbp * std::real(256.0 / prod);     // 4^order / prod(4 - p)
  int n = 0;
  for (int i = 0; i < 8 && n < 4; ++i) {
    std::complex<double> pd = (4.0 + pbp[i]) / (4.0 - pbp[i]);
    if (std::imag(pd) > 0.0) {
      double g = (n == 0) ? kd : 1.0;
      out->b0[n] = (float)g;
      out->b2[n] = (float)(-g);
      out->a1[n] = (float)(-2.0 * std::real(pd));
      out->a2[n] = (float)std::norm(pd);
      ++n;
    }
  }
}

// ---------------- kernel 1: chunk-parallel IIR (DF2T, 4 biquads), write f16 padded ----
// grid: (22 seq-blocks, NCHK chunks, nbands). blockIdx.z selects xf slot; band =
// band_base + blockIdx.z selects coefficients.
__global__ void st_iir_kernel(const float* __restrict__ x, _Float16* __restrict__ xf,
                              Sos3 coef, int band_base) {
  const int seq   = blockIdx.x * blockDim.x + threadIdx.x;   // (b*C + c)
  if (seq >= B_ * C_) return;
  const int chunk = blockIdx.y;
  const int band  = band_base + blockIdx.z;
  const Sos sos   = coef.s[band];

  const float* xin = x + (size_t)seq * T_;
  _Float16*    xo  = xf + ((size_t)blockIdx.z * (B_ * C_) + seq) * TPAD;

  const int wstart = chunk * PCHUNK;
  const int wend   = (wstart + PCHUNK < T_) ? (wstart + PCHUNK) : T_;
  const int tstart = (wstart > WARM) ? (wstart - WARM) : 0;

  if (chunk == 0)        for (int t = 0; t < STR; ++t)         xo[t] = (_Float16)0.0f;
  if (chunk == NCHK - 1) for (int t = STR + T_; t < TPAD; ++t) xo[t] = (_Float16)0.0f;

  float z1[4] = {0.f, 0.f, 0.f, 0.f}, z2[4] = {0.f, 0.f, 0.f, 0.f};
  for (int t = tstart; t < wstart; ++t) {          // warm-up: no stores
    float cur = xin[t];
#pragma unroll
    for (int s = 0; s < 4; ++s) {                  // b1 == 0 for all sections
      float y = sos.b0[s] * cur + z1[s];
      z1[s]   = z2[s] - sos.a1[s] * y;
      z2[s]   = sos.b2[s] * cur - sos.a2[s] * y;
      cur = y;
    }
  }
  for (int t = wstart; t < wend; ++t) {            // payload: store f16
    float cur = xin[t];
#pragma unroll
    for (int s = 0; s < 4; ++s) {
      float y = sos.b0[s] * cur + z1[s];
      z1[s]   = z2[s] - sos.a1[s] * y;
      z2[s]   = sos.b2[s] * cur - sos.a2[s] * y;
      cur = y;
    }
    xo[STR + t] = (_Float16)cur;
  }
}

// ---------------- kernel 2: pack weights into WMMA B-operand layout ----------------
// bp[((nt*KCH + chunk)*32 + lane)*16 + h] ; lane holds column d = nt*16 + lane%16,
// halves 0..7 -> k_local = 8*(lane/16)+h ; halves 8..15 -> k_local = 8*(lane/16)+16+(h-8)
__global__ void st_prep_w_kernel(const float* __restrict__ w, _Float16* __restrict__ bp) {
  int idx = blockIdx.x * blockDim.x + threadIdx.x;
  if (idx >= 4 * KCH * 32 * 16) return;
  int h     = idx & 15;
  int lane  = (idx >> 4) & 31;
  int t9    = idx >> 9;
  int chunk = t9 % KCH;
  int nt    = t9 / KCH;
  int d     = nt * 16 + (lane & 15);
  int ko    = (lane >> 4) * 8;
  int kl    = (h < 8) ? (ko + h) : (ko + 16 + (h - 8));
  int k     = chunk * 32 + kl;
  int c     = k >> 6;
  int wi    = k & 63;
  float v   = (wi < WIN) ? w[((size_t)d * C_ + c) * WIN + wi] : 0.0f;
  bp[idx]   = (_Float16)v;
}

// ---------------- kernel 3: implicit-GEMM conv via v_wmma_f32_16x16x32_f16 ----------------
// M = B*L rows, N = 64, K = 44 chunks of 32. 4 waves/WG, each wave: 16x64 tile (4 accs).
__global__ void __launch_bounds__(128) st_conv_wmma_kernel(const _Float16* __restrict__ xf,
                                                           const _Float16* __restrict__ bp,
                                                           float* __restrict__ y) {
  const int lane   = threadIdx.x & 31;
  const int wave   = threadIdx.x >> 5;
  const int tile_m = blockIdx.x * 64 + wave * 16;
  const int mrow   = tile_m + (lane & 15);     // A-layout: lane holds row M = lane%16
  const int b      = mrow / L_;
  const int l      = mrow - b * L_;
  const int hi     = lane >> 4;                // 0: K {0-7,16-23}; 1: K {8-15,24-31}
  const int ko     = hi * 8;
  const _Float16* arow = xf + (size_t)(b * C_) * TPAD;

  v8f acc0 = {}; v8f acc1 = {}; v8f acc2 = {}; v8f acc3 = {};

  for (int chunk = 0; chunk < KCH; ++chunk) {
    const int c  = chunk >> 1;
    const int wb = (chunk & 1) * 32;
    const _Float16* ap = arow + (size_t)c * TPAD + l * STR + wb + ko;  // xf idx = l*31 + w
    v16h a;
    ((v8h_u*)&a)[0] = *(const v8h_u*)(ap);
    ((v8h_u*)&a)[1] = *(const v8h_u*)(ap + 16);

    const _Float16* bpc = bp + ((size_t)chunk * 32 + lane) * 16;
    v16h bm0, bm1, bm2, bm3;
    {
      const _Float16* p = bpc;                                    // nt = 0
      ((v8h*)&bm0)[0] = *(const v8h*)(p); ((v8h*)&bm0)[1] = *(const v8h*)(p + 8);
      p = bpc + (size_t)1 * KCH * 32 * 16;                        // nt = 1
      ((v8h*)&bm1)[0] = *(const v8h*)(p); ((v8h*)&bm1)[1] = *(const v8h*)(p + 8);
      p = bpc + (size_t)2 * KCH * 32 * 16;                        // nt = 2
      ((v8h*)&bm2)[0] = *(const v8h*)(p); ((v8h*)&bm2)[1] = *(const v8h*)(p + 8);
      p = bpc + (size_t)3 * KCH * 32 * 16;                        // nt = 3
      ((v8h*)&bm3)[0] = *(const v8h*)(p); ((v8h*)&bm3)[1] = *(const v8h*)(p + 8);
    }
    acc0 = __builtin_amdgcn_wmma_f32_16x16x32_f16(false, a, false, bm0, (short)0, acc0, false, false);
    acc1 = __builtin_amdgcn_wmma_f32_16x16x32_f16(false, a, false, bm1, (short)0, acc1, false, false);
    acc2 = __builtin_amdgcn_wmma_f32_16x16x32_f16(false, a, false, bm2, (short)0, acc2, false, false);
    acc3 = __builtin_amdgcn_wmma_f32_16x16x32_f16(false, a, false, bm3, (short)0, acc3, false, false);
  }

  // C/D layout: VGPR r, lane L -> row M = r + 8*(L/16), col N = L%16. Store y[m][d].
  const int dcol = lane & 15;
#pragma unroll
  for (int r = 0; r < 8; ++r) {
    const size_t m = (size_t)(tile_m + r + 8 * hi);
    y[m * 64 + 0  + dcol] = acc0[r];
    y[m * 64 + 16 + dcol] = acc1[r];
    y[m * 64 + 32 + dcol] = acc2[r];
    y[m * 64 + 48 + dcol] = acc3[r];
  }
}

// ---------------- kernel 4a: zero stats ----------------
__global__ void st_zero_kernel(float* __restrict__ s) {
  if (threadIdx.x < 128) s[threadIdx.x] = 0.0f;
}

// ---------------- kernel 4b: BN batch statistics (sum, sumsq per channel) ----------------
__global__ void st_stats_kernel(const float* __restrict__ y, float* __restrict__ stats) {
  const int d    = threadIdx.x & 63;
  const int row0 = blockIdx.x * 256 + (threadIdx.x >> 6) * 64;
  float s1 = 0.0f, s2 = 0.0f;
  for (int i = 0; i < 64; ++i) {
    float v = y[(size_t)(row0 + i) * 64 + d];
    s1 += v; s2 += v * v;
  }
  atomicAdd(&stats[d],      s1);
  atomicAdd(&stats[64 + d], s2);
}

// ---------------- kernel 5: BN + exact GELU + pos embed, write [B,L,D] slice ----------------
__global__ void st_final_kernel(const float* __restrict__ y, const float* __restrict__ stats,
                                const float* __restrict__ gamma, const float* __restrict__ beta,
                                const float* __restrict__ pos, float* __restrict__ out) {
  size_t idx = (size_t)blockIdx.x * 256 + threadIdx.x;
  if (idx >= (size_t)MT * 64) return;
  const int d = (int)(idx & 63);
  const int m = (int)(idx >> 6);
  const int l = m % L_;
  const float inv_n = 1.0f / (float)MT;
  float mean = stats[d] * inv_n;
  float var  = stats[64 + d] * inv_n - mean * mean;
  float sc   = gamma[d] * rsqrtf(var + BN_EPS);
  float sh   = beta[d] - mean * sc;
  float v    = y[idx] * sc + sh;
  float g    = 0.5f * v * (1.0f + erff(v * 0.70710678118654752f));   // exact GELU
  out[idx]   = g + pos[l * 64 + d];
}

// ---------------- launcher ----------------
extern "C" void kernel_launch(void* const* d_in, const int* in_sizes, int n_in,
                              void* d_out, int out_size, void* d_ws, size_t ws_size,
                              hipStream_t stream) {
  const float* x      = (const float*)d_in[0];   // [128,22,5001]
  const float* conv_w = (const float*)d_in[1];   // [3,64,22,62]
  const float* gamma  = (const float*)d_in[2];   // [3,64]
  const float* beta   = (const float*)d_in[3];   // [3,64]
  const float* pos    = (const float*)d_in[4];   // [1,162,64]
  float*       out    = (float*)d_out;           // [3,128,162,64]

  const size_t XFSZ = (size_t)B_ * C_ * TPAD * sizeof(_Float16);   // 28.8 MB
  const size_t YSZ  = (size_t)MT * 64 * sizeof(float);             //  5.3 MB
  const size_t BPSZ = (size_t)4 * KCH * 32 * 16 * sizeof(_Float16);// 180 KB
  auto align256 = [](size_t v) { return (v + 255) & ~(size_t)255; };

  const size_t need_par = align256(NB * XFSZ) + align256(YSZ) + align256(NB * BPSZ) + 512;
  const bool parallel_bands = (ws_size >= need_par);
  const int  nslots = parallel_bands ? NB : 1;

  char* ws = (char*)d_ws;
  _Float16* xf    = (_Float16*)ws;                                  // nslots * 28.8 MB
  size_t off      = align256(nslots * XFSZ);
  float* ybuf     = (float*)(ws + off);  off  = align256(off + YSZ);
  _Float16* bpack = (_Float16*)(ws + off); off = align256(off + nslots * BPSZ);
  float* stats    = (float*)(ws + off);

  Sos3 coef;
  butter_sos(4.0,  8.0,  &coef.s[0]);   // theta
  butter_sos(8.0,  13.0, &coef.s[1]);   // mu
  butter_sos(13.0, 30.0, &coef.s[2]);   // beta

  const int seq_blocks  = (B_ * C_ + 127) / 128;                    // 22
  const int prep_blocks = (4 * KCH * 32 * 16 + 255) / 256;

  if (parallel_bands) {
    // one launch covers all bands x chunks: 3*10*2816 threads = 2640 waves
    st_iir_kernel<<<dim3(seq_blocks, NCHK, NB), 128, 0, stream>>>(x, xf, coef, 0);
    for (int band = 0; band < NB; ++band)
      st_prep_w_kernel<<<prep_blocks, 256, 0, stream>>>(
          conv_w + (size_t)band * D_ * C_ * WIN, bpack + (size_t)band * (BPSZ / 2));
  }

  for (int band = 0; band < NB; ++band) {
    const _Float16* xf_b = xf    + (size_t)(parallel_bands ? band : 0) * (XFSZ / 2);
    const _Float16* bp_b = bpack + (size_t)(parallel_bands ? band : 0) * (BPSZ / 2);

    if (!parallel_bands) {
      st_iir_kernel<<<dim3(seq_blocks, NCHK, 1), 128, 0, stream>>>(x, xf, coef, band);
      st_prep_w_kernel<<<prep_blocks, 256, 0, stream>>>(
          conv_w + (size_t)band * D_ * C_ * WIN, bpack);
    }

    st_zero_kernel<<<1, 128, 0, stream>>>(stats);

    st_conv_wmma_kernel<<<MT / 64, 128, 0, stream>>>(xf_b, bp_b, ybuf);

    st_stats_kernel<<<MT / 256, 256, 0, stream>>>(ybuf, stats);

    st_final_kernel<<<((size_t)MT * 64 + 255) / 256, 256, 0, stream>>>(
        ybuf, stats, gamma + band * D_, beta + band * D_, pos,
        out + (size_t)band * MT * 64);
  }
}
